// MultiHeadAttention_90941637525709
// MI455X (gfx1250) — matrix operand order repaired and checked by reading, not compiled, not run on hardware
//
#include <hip/hip_runtime.h>
#include <hip/hip_bf16.h>

// MI455X (gfx1250) fused multi-head attention, bf16 WMMA pipeline.
// x:[2,2048,1024] f32; Wq/Wk/Wv/Wo:[1024,1024] f32; bo:[1024] f32 -> out f32.

#define EMB   1024
#define LSEQ  2048
#define NBATCH 2
#define NHEAD 16
#define HDIM  64
#define MTOT  (NBATCH * LSEQ)   // 4096

typedef __attribute__((ext_vector_type(16))) __bf16 bf16x16;
typedef __attribute__((ext_vector_type(8)))  __bf16 bf16x8;
typedef __attribute__((ext_vector_type(4)))  __bf16 bf16x4;
typedef __attribute__((ext_vector_type(8)))  float  f32x8;
typedef __attribute__((ext_vector_type(4)))  float  f32x4;

__device__ __forceinline__ f32x8 f32x8_zero() {
    f32x8 v;
#pragma unroll
    for (int i = 0; i < 8; ++i) v[i] = 0.0f;
    return v;
}

__device__ __forceinline__ f32x8 wmma_bf16(bf16x16 a, bf16x16 b, f32x8 c) {
    // v_wmma_f32_16x16x32_bf16: D = A(16x32) * B(32x16) + C
    return __builtin_amdgcn_wmma_f32_16x16x32_bf16(
        /*neg_a=*/false, a, /*neg_b=*/false, b,
        /*c_mod=*/(short)0, c, /*reuse_a=*/false, /*reuse_b=*/false);
}

// A-matrix 16x32 bf16 fragment (ISA 7.12.2): lane = row (lane&15),
// per-lane K chunks: [k + hi*8 .. +7] and [k + hi*8 + 16 .. +7]  (hi = lane>>4)
__device__ __forceinline__ bf16x16 load_a_frag(const __bf16* rowp, int k, int hi) {
    union { bf16x16 v; bf16x8 h[2]; } u;
    u.h[0] = *(const bf16x8*)(rowp + k + hi * 8);
    u.h[1] = *(const bf16x8*)(rowp + k + hi * 8 + 16);
    return u.v;
}

// B-matrix 32x16 bf16 fragment: lane = column (lane&15), per-lane K chunks
// [k + hi*8 .. +7] and [k + hi*8 + 16 .. +7] (same K order as A fragment).
__device__ __forceinline__ bf16x16 load_b_frag(const __bf16* colp, int k, int hi) {
    union { bf16x16 v; bf16x8 h[2]; } u;
    u.h[0] = *(const bf16x8*)(colp + k + hi * 8);
    u.h[1] = *(const bf16x8*)(colp + k + hi * 8 + 16);
    return u.v;
}

// ---------------- f32 -> bf16 conversion (vectorized x4) ----------------
__global__ __launch_bounds__(256)
void cvt_f32_bf16(const float* __restrict__ src, __bf16* __restrict__ dst, int n4) {
    int i = blockIdx.x * blockDim.x + threadIdx.x;
    if (i < n4) {
        f32x4 v = *(const f32x4*)(src + (size_t)i * 4);
        bf16x4 o;
#pragma unroll
        for (int j = 0; j < 4; ++j) o[j] = (__bf16)v[j];
        *(bf16x4*)(dst + (size_t)i * 4) = o;
    }
}

// ---------------- projection GEMM: C[M,E] = X[M,E] * W[E,E]^T --------------
// block = 128 threads (4 waves); each wave computes a 32x64 tile.
// grid = (MTOT/32, EMB/256).  transposed=1 stores V^T layout [nb][col][l].
__global__ __launch_bounds__(128)
void gemm_proj_bf16(const __bf16* __restrict__ X, const __bf16* __restrict__ W,
                    __bf16* __restrict__ out, int transposed) {
    __shared__ __attribute__((aligned(16))) __bf16 stg[4][16 * 16]; // per-wave store stage

    const int lane = threadIdx.x & 31;
    const int wave = threadIdx.x >> 5;
    const int r  = lane & 15;
    const int hi = lane >> 4;
    const int row_base = blockIdx.x * 32;
    const int col_base = blockIdx.y * 256 + wave * 64;

    f32x8 acc[2][4];
#pragma unroll
    for (int a = 0; a < 2; ++a)
#pragma unroll
        for (int b = 0; b < 4; ++b) acc[a][b] = f32x8_zero();

    const __bf16* arow0 = X + (size_t)(row_base + r) * EMB;
    const __bf16* arow1 = arow0 + (size_t)16 * EMB;
    const __bf16* brow[4];
#pragma unroll
    for (int nt = 0; nt < 4; ++nt)
        brow[nt] = W + (size_t)(col_base + nt * 16 + r) * EMB;

    for (int k = 0; k < EMB; k += 32) {
        // distinct registers for all fragments -> one big load clause,
        // staggered s_wait_loadcnt instead of full serialization
        bf16x16 a0 = load_a_frag(arow0, k, hi);
        bf16x16 a1 = load_a_frag(arow1, k, hi);
        bf16x16 bf[4];
#pragma unroll
        for (int nt = 0; nt < 4; ++nt) bf[nt] = load_b_frag(brow[nt], k, hi);
        if (k + 32 < EMB) {
            // locality=3 -> nearest-scope prefetch (pull into WGP$ + L2)
            __builtin_prefetch(arow0 + k + 32, 0, 3);
            __builtin_prefetch(arow1 + k + 32, 0, 3);
        }
#pragma unroll
        for (int nt = 0; nt < 4; ++nt) {
            acc[0][nt] = wmma_bf16(a0, bf[nt], acc[0][nt]);
            acc[1][nt] = wmma_bf16(a1, bf[nt], acc[1][nt]);
        }
    }

    if (!transposed) {
        // transpose each 16x16 tile through per-wave LDS (DS ops are in-order
        // within a wave: no barrier needed) -> one packed b128 store per tile.
#pragma unroll
        for (int mi = 0; mi < 2; ++mi)
#pragma unroll
            for (int nt = 0; nt < 4; ++nt) {
#pragma unroll
                for (int j = 0; j < 8; ++j)
                    stg[wave][(hi * 8 + j) * 16 + r] = (__bf16)acc[mi][nt][j];
                bf16x8 rowv = *(const bf16x8*)&stg[wave][r * 16 + hi * 8];
                *(bf16x8*)(out + (size_t)(row_base + mi * 16 + r) * EMB
                               + col_base + nt * 16 + hi * 8) = rowv;
            }
    } else {
        // transposed store: out[(nb*EMB + col) * LSEQ + l] -- per-lane rows are
        // 8 consecutive l values -> one packed 16B store per fragment.
        int nb = row_base >> 11;                 // row tiles never cross batch
        int lbase = (row_base & (LSEQ - 1));
#pragma unroll
        for (int mi = 0; mi < 2; ++mi)
#pragma unroll
            for (int nt = 0; nt < 4; ++nt) {
                int col = col_base + nt * 16 + r;
                int l0 = lbase + mi * 16 + hi * 8;
                bf16x8 st;
#pragma unroll
                for (int j = 0; j < 8; ++j) st[j] = (__bf16)acc[mi][nt][j];
                *(bf16x8*)(out + ((size_t)(nb * EMB + col)) * LSEQ + l0) = st;
            }
    }
}

// ---------------- fused flash attention -----------------------------------
// grid = (LSEQ/64, NHEAD, NBATCH), block = 128 (4 waves); each wave owns a
// 16-query tile, streams 32 keys per iteration with online softmax.
__global__ __launch_bounds__(128)
void attention_fused(const __bf16* __restrict__ Qb, const __bf16* __restrict__ Kb,
                     const __bf16* __restrict__ Vt, __bf16* __restrict__ Ob) {
    __shared__ __attribute__((aligned(16))) __bf16 lds[4][16 * 32]; // P staging per wave

    const int lane = threadIdx.x & 31;
    const int wave = threadIdx.x >> 5;
    const int r  = lane & 15;
    const int hi = lane >> 4;
    const int q0 = (blockIdx.x * 4 + wave) * 16;
    const int h  = blockIdx.y;
    const int nb = blockIdx.z;
    const float scale = 0.03125f;                 // 1/sqrt(EMB) = 1/32

    // Q tile 16x64 as two 16x32 A fragments (held for the whole key loop)
    const __bf16* qrow = Qb + (size_t)(nb * LSEQ + q0 + r) * EMB + h * HDIM;
    bf16x16 qa0 = load_a_frag(qrow, 0, hi);
    bf16x16 qa1 = load_a_frag(qrow, 32, hi);

    const __bf16* kbase = Kb + (size_t)(nb * LSEQ + r) * EMB + h * HDIM;
    const __bf16* vbase[4];
#pragma unroll
    for (int t = 0; t < 4; ++t)
        vbase[t] = Vt + (size_t)(nb * EMB + h * HDIM + t * 16 + r) * LSEQ;

    f32x8 o[4];
#pragma unroll
    for (int t = 0; t < 4; ++t) o[t] = f32x8_zero();
    float mrow[8], lrow[8];
#pragma unroll
    for (int j = 0; j < 8; ++j) { mrow[j] = -1e30f; lrow[j] = 0.0f; }

    for (int kb = 0; kb < LSEQ; kb += 32) {
        // ---- issue K fragment loads (4) then V fragment loads (4): the S
        // WMMAs only need loadcnt<=4, so V stays in flight under the softmax.
        bf16x16 kf[2][2];
#pragma unroll
        for (int kt = 0; kt < 2; ++kt) {
            const __bf16* krow = kbase + (size_t)(kb + kt * 16) * EMB;
            kf[kt][0] = load_b_frag(krow, 0, hi);
            kf[kt][1] = load_b_frag(krow, 32, hi);
        }
        bf16x16 vb[4];
#pragma unroll
        for (int t = 0; t < 4; ++t) vb[t] = load_b_frag(vbase[t], kb, hi);

        if (kb + 32 < LSEQ) {   // global_prefetch_b8 next key block (near scope)
            __builtin_prefetch(kbase + (size_t)(kb + 32) * EMB, 0, 3);
            __builtin_prefetch(kbase + (size_t)(kb + 48) * EMB, 0, 3);
            __builtin_prefetch(vbase[0] + kb + 32, 0, 3);
            __builtin_prefetch(vbase[2] + kb + 32, 0, 3);
        }

        // ---- S = Q K^T for two 16-key tiles (K-dim = HDIM = 2 x 32) ----
        f32x8 s[2];
#pragma unroll
        for (int kt = 0; kt < 2; ++kt) {
            f32x8 acc = f32x8_zero();
            acc = wmma_bf16(qa0, kf[kt][0], acc);
            acc = wmma_bf16(qa1, kf[kt][1], acc);
            s[kt] = acc;
        }
        // ---- online softmax: row max / exp / row sum (16-lane butterflies) ----
#pragma unroll
        for (int j = 0; j < 8; ++j) {
            float v0 = s[0][j] * scale;
            float v1 = s[1][j] * scale;
            s[0][j] = v0; s[1][j] = v1;
            float v = fmaxf(v0, v1);
            v = fmaxf(v, __shfl_xor(v, 1));
            v = fmaxf(v, __shfl_xor(v, 2));
            v = fmaxf(v, __shfl_xor(v, 4));
            v = fmaxf(v, __shfl_xor(v, 8));
            float mn = fmaxf(mrow[j], v);
            float cf = __expf(mrow[j] - mn);
            mrow[j] = mn;
            float p0 = __expf(s[0][j] - mn);
            float p1 = __expf(s[1][j] - mn);
            s[0][j] = p0; s[1][j] = p1;
            float rs = p0 + p1;
            rs += __shfl_xor(rs, 1);
            rs += __shfl_xor(rs, 2);
            rs += __shfl_xor(rs, 4);
            rs += __shfl_xor(rs, 8);
            lrow[j] = lrow[j] * cf + rs;
#pragma unroll
            for (int t = 0; t < 4; ++t) o[t][j] *= cf;
        }
        // ---- transpose P (C-layout -> A-layout) through LDS ----
#pragma unroll
        for (int kt = 0; kt < 2; ++kt)
#pragma unroll
            for (int j = 0; j < 8; ++j)
                lds[wave][(j + hi * 8) * 32 + kt * 16 + r] = (__bf16)s[kt][j];
        __syncthreads();
        bf16x16 pa = load_a_frag(&lds[wave][r * 32], 0, hi);
        // ---- O += P * V (V fragments already resident) ----
#pragma unroll
        for (int t = 0; t < 4; ++t) o[t] = wmma_bf16(pa, vb[t], o[t]);
        __syncthreads();
    }

    // normalize; transpose each 16x16 output tile through LDS -> b128 stores
    float rinv[8];
#pragma unroll
    for (int j = 0; j < 8; ++j) rinv[j] = 1.0f / lrow[j];
#pragma unroll
    for (int t = 0; t < 4; ++t) {
#pragma unroll
        for (int j = 0; j < 8; ++j)
            lds[wave][(hi * 8 + j) * 16 + r] = (__bf16)(o[t][j] * rinv[j]);
        bf16x8 rowv = *(const bf16x8*)&lds[wave][r * 16 + hi * 8];
        *(bf16x8*)(Ob + (size_t)(nb * LSEQ + q0 + r) * EMB
                       + h * HDIM + t * 16 + hi * 8) = rowv;
    }
}

// ---------------- output projection: out = A * Wo^T + bo (f32 out) --------
__global__ __launch_bounds__(128)
void gemm_out_f32(const __bf16* __restrict__ X, const __bf16* __restrict__ W,
                  const float* __restrict__ bias, float* __restrict__ out) {
    __shared__ __attribute__((aligned(16))) float stgf[4][16 * 16]; // per-wave stage

    const int lane = threadIdx.x & 31;
    const int wave = threadIdx.x >> 5;
    const int r  = lane & 15;
    const int hi = lane >> 4;
    const int row_base = blockIdx.x * 32;
    const int col_base = blockIdx.y * 256 + wave * 64;

    f32x8 acc[2][4];
#pragma unroll
    for (int a = 0; a < 2; ++a)
#pragma unroll
        for (int b = 0; b < 4; ++b) acc[a][b] = f32x8_zero();

    const __bf16* arow0 = X + (size_t)(row_base + r) * EMB;
    const __bf16* arow1 = arow0 + (size_t)16 * EMB;
    const __bf16* brow[4];
#pragma unroll
    for (int nt = 0; nt < 4; ++nt)
        brow[nt] = W + (size_t)(col_base + nt * 16 + r) * EMB;

    for (int k = 0; k < EMB; k += 32) {
        bf16x16 a0 = load_a_frag(arow0, k, hi);
        bf16x16 a1 = load_a_frag(arow1, k, hi);
        bf16x16 bf[4];
#pragma unroll
        for (int nt = 0; nt < 4; ++nt) bf[nt] = load_b_frag(brow[nt], k, hi);
        if (k + 32 < EMB) {
            __builtin_prefetch(arow0 + k + 32, 0, 3);
            __builtin_prefetch(arow1 + k + 32, 0, 3);
        }
#pragma unroll
        for (int nt = 0; nt < 4; ++nt) {
            acc[0][nt] = wmma_bf16(a0, bf[nt], acc[0][nt]);
            acc[1][nt] = wmma_bf16(a1, bf[nt], acc[1][nt]);
        }
    }

    // add bias, transpose 16x16 f32 tiles through LDS -> two b128 stores/row
#pragma unroll
    for (int nt = 0; nt < 4; ++nt) {
        float bcol = bias[col_base + nt * 16 + r];
#pragma unroll
        for (int mi = 0; mi < 2; ++mi) {
#pragma unroll
            for (int j = 0; j < 8; ++j)
                stgf[wave][(hi * 8 + j) * 16 + r] = acc[mi][nt][j] + bcol;
            f32x4 r0 = *(const f32x4*)&stgf[wave][r * 16 + hi * 8];
            f32x4 r1 = *(const f32x4*)&stgf[wave][r * 16 + hi * 8 + 4];
            float* op = out + (size_t)(row_base + mi * 16 + r) * EMB
                            + col_base + nt * 16 + hi * 8;
            *(f32x4*)op       = r0;
            *(f32x4*)(op + 4) = r1;
        }
    }
}

extern "C" void kernel_launch(void* const* d_in, const int* in_sizes, int n_in,
                              void* d_out, int out_size, void* d_ws, size_t ws_size,
                              hipStream_t stream) {
    const float* x  = (const float*)d_in[0];
    const float* Wq = (const float*)d_in[1];
    const float* Wk = (const float*)d_in[2];
    const float* Wv = (const float*)d_in[3];
    const float* Wo = (const float*)d_in[4];
    const float* bo = (const float*)d_in[5];

    // workspace layout (bf16 elements), 48 MB total
    __bf16* ws  = (__bf16*)d_ws;
    __bf16* xb  = ws;                               // 4096*1024
    __bf16* wqb = xb  + (size_t)MTOT * EMB;         // 1024*1024
    __bf16* wkb = wqb + (size_t)EMB * EMB;
    __bf16* wvb = wkb + (size_t)EMB * EMB;
    __bf16* wob = wvb + (size_t)EMB * EMB;
    __bf16* Qb  = wob + (size_t)EMB * EMB;          // 4096*1024
    __bf16* Kb  = Qb  + (size_t)MTOT * EMB;
    __bf16* Vt  = Kb  + (size_t)MTOT * EMB;         // transposed [nb][col][l]
    __bf16* Ab  = Vt  + (size_t)MTOT * EMB;         // attention output

    const int nx4 = (MTOT * EMB) / 4;               // 1048576
    const int nw4 = (EMB * EMB) / 4;                // 262144
    cvt_f32_bf16<<<nx4 / 256, 256, 0, stream>>>(x,  xb,  nx4);
    cvt_f32_bf16<<<nw4 / 256, 256, 0, stream>>>(Wq, wqb, nw4);
    cvt_f32_bf16<<<nw4 / 256, 256, 0, stream>>>(Wk, wkb, nw4);
    cvt_f32_bf16<<<nw4 / 256, 256, 0, stream>>>(Wv, wvb, nw4);
    cvt_f32_bf16<<<nw4 / 256, 256, 0, stream>>>(Wo, wob, nw4);

    dim3 ggrid(MTOT / 32, EMB / 256);               // (128, 4)
    gemm_proj_bf16<<<ggrid, 128, 0, stream>>>(xb, wqb, Qb, 0);
    gemm_proj_bf16<<<ggrid, 128, 0, stream>>>(xb, wkb, Kb, 0);
    gemm_proj_bf16<<<ggrid, 128, 0, stream>>>(xb, wvb, Vt, 1);

    attention_fused<<<dim3(LSEQ / 64, NHEAD, NBATCH), 128, 0, stream>>>(Qb, Kb, Vt, Ab);

    gemm_out_f32<<<ggrid, 128, 0, stream>>>(Ab, wob, bo, (float*)d_out);
}
